// DiffTime_24326694764683
// MI455X (gfx1250) — compile-verified, matrix operand order checked by reading, not run on hardware
//
#include <hip/hip_runtime.h>
#include <hip/hip_bf16.h>

// ---------------- problem constants ----------------
#define NB     16384      // batch
#define NEMB   300        // embedding dim
#define NEMBP  320        // padded K (10 chunks of 32)
#define NH     100        // H1 == H2 == PROD
#define NHP    112        // padded h (7 tiles of 16)
#define NPROD  100
#define KCHUNKS 10        // 320 / 32

typedef __attribute__((ext_vector_type(16))) __bf16 v16bf;
typedef __attribute__((ext_vector_type(8)))  __bf16 v8bf;
typedef __attribute__((ext_vector_type(8)))  float  v8f;

__device__ __forceinline__ unsigned short f2bf(float x) {
  unsigned int u = __float_as_uint(x);
  unsigned int r = u + 0x7FFFu + ((u >> 16) & 1u);   // round-to-nearest-even
  return (unsigned short)(r >> 16);
}

// ---------------- K0a: h1[b,i] = tanh(t[b]*h1_k[i] + h1_b[i]) ----------------
__global__ void k_h1(const float* __restrict__ times, const float* __restrict__ h1_k,
                     const float* __restrict__ h1_b, float* __restrict__ h1w) {
  int i = blockIdx.x * 256 + threadIdx.x;
  if (i >= NB * NH) return;
  int b = i / NH, j = i % NH;
  h1w[i] = tanhf(times[b] * h1_k[j] + h1_b[j]);
}

// ---------------- K0b: tv[b,h] = tanh(h1[b,:] @ h2_k[:,h] + h2_b[h]), padded to 112 ----------------
__global__ __launch_bounds__(128) void k_tv(const float* __restrict__ h1w,
                                            const float* __restrict__ h2_k,
                                            const float* __restrict__ h2_b,
                                            float* __restrict__ tv) {
  __shared__ float h1s[NH];
  int b = blockIdx.x, h = threadIdx.x;
  if (h < NH) h1s[h] = h1w[(size_t)b * NH + h];
  __syncthreads();
  if (h < NHP) {
    float out = 0.f;
    if (h < NH) {
      float acc = h2_b[h];
      for (int i = 0; i < NH; ++i) acc += h1s[i] * h2_k[i * NH + h];
      out = tanhf(acc);
    }
    tv[(size_t)b * NHP + h] = out;
  }
}

// ---------------- K1a: gather embeddings -> bf16 [2][B][320] ----------------
__global__ void k_gather(const int* __restrict__ idxT, const int* __restrict__ idxC,
                         const float* __restrict__ tabT, const float* __restrict__ tabC,
                         unsigned short* __restrict__ embB) {
  size_t i = (size_t)blockIdx.x * 256 + threadIdx.x;
  if (i >= (size_t)2 * NB * NEMBP) return;
  int e = (int)(i % NEMBP);
  size_t rem = i / NEMBP;
  int b = (int)(rem % NB);
  int br = (int)(rem / NB);
  const int*   idx = br ? idxC : idxT;
  const float* tab = br ? tabC : tabT;
  float v = (e < NEMB) ? tab[(size_t)idx[b] * NEMB + e] : 0.0f;
  embB[i] = f2bf(v);
}

// ---------------- K1b: Bp[p][h][e] = evoke_k[e][p*100+h] (bf16, zero padded) ----------------
__global__ void k_evoke(const float* __restrict__ evoke_k, unsigned short* __restrict__ Bp) {
  size_t i = (size_t)blockIdx.x * 256 + threadIdx.x;
  if (i >= (size_t)NPROD * NHP * NEMBP) return;
  int e = (int)(i % NEMBP);
  size_t rem = i / NEMBP;
  int h = (int)(rem % NHP);
  int p = (int)(rem / NHP);
  float v = (e < NEMB && h < NH) ? evoke_k[(size_t)e * (NH * NPROD) + p * NH + h] : 0.0f;
  Bp[i] = f2bf(v);
}

// ---------------- K2: fused GEMM + tv contraction (the 197-GFLOP kernel) ----------------
// Both branches computed in the same wave so every B operand feeds 2 WMMAs:
// mv[br][b][p] = sum_h ( sum_e emb[br][b][e]*evoke_k[e][p*100+h] + evoke_b[p*100+h] ) * tv[b][h]
#define BT 64   // b-rows per workgroup (4 waves x 16)
__global__ __launch_bounds__(128) void k_main(const unsigned short* __restrict__ embB, // [2][B][320]
                                              const unsigned short* __restrict__ Bp,   // [100][112][320]
                                              const float* __restrict__ tv,            // [B][112]
                                              const float* __restrict__ evoke_b,       // [10000]
                                              float* __restrict__ mv) {                 // [2][B][100]
  __shared__ float tv_lds[BT][NHP];
  __shared__ float bias_lds[NHP];

  const int b0  = blockIdx.x * BT;
  const int p   = blockIdx.y;
  const int tid = threadIdx.x;

  for (int i = tid; i < BT * NHP; i += 128) {
    int r = i / NHP, hh = i % NHP;
    tv_lds[r][hh] = tv[(size_t)(b0 + r) * NHP + hh];
  }
  if (tid < NHP) bias_lds[tid] = (tid < NH) ? evoke_b[p * NH + tid] : 0.0f;
  __syncthreads();

  const int wave = tid >> 5;
  const int lane = tid & 31;
  const int lrow = lane & 15;
  const int ko   = (lane >> 4) * 8;   // 16-bit operand K sub-chunk offset per half-wave

  const unsigned short* ArowT = embB + ((size_t)0 * NB + (b0 + wave * 16 + lrow)) * NEMBP;
  const unsigned short* ArowC = embB + ((size_t)1 * NB + (b0 + wave * 16 + lrow)) * NEMBP;
  const unsigned short* Bbas  = Bp + (size_t)p * NHP * NEMBP;

  v8f accT[7], accC[7];
#pragma unroll
  for (int t = 0; t < 7; ++t) {
    accT[t] = (v8f){0, 0, 0, 0, 0, 0, 0, 0};
    accC[t] = (v8f){0, 0, 0, 0, 0, 0, 0, 0};
  }

  for (int kc = 0; kc < KCHUNKS; ++kc) {
    const unsigned short* apT = ArowT + kc * 32 + ko;
    const unsigned short* apC = ArowC + kc * 32 + ko;
    v8bf aTlo = *(const v8bf*)(apT);
    v8bf aThi = *(const v8bf*)(apT + 16);
    v8bf aClo = *(const v8bf*)(apC);
    v8bf aChi = *(const v8bf*)(apC + 16);
    v16bf aT = __builtin_shufflevector(aTlo, aThi, 0, 1, 2, 3, 4, 5, 6, 7,
                                       8, 9, 10, 11, 12, 13, 14, 15);
    v16bf aC = __builtin_shufflevector(aClo, aChi, 0, 1, 2, 3, 4, 5, 6, 7,
                                       8, 9, 10, 11, 12, 13, 14, 15);
#pragma unroll
    for (int t = 0; t < 7; ++t) {
      const unsigned short* bp = Bbas + (size_t)(t * 16 + lrow) * NEMBP + kc * 32 + ko;
      v8bf blo = *(const v8bf*)(bp);
      v8bf bhi = *(const v8bf*)(bp + 16);
      v16bf bm = __builtin_shufflevector(blo, bhi, 0, 1, 2, 3, 4, 5, 6, 7,
                                         8, 9, 10, 11, 12, 13, 14, 15);
      accT[t] = __builtin_amdgcn_wmma_f32_16x16x32_bf16(
          false, aT, false, bm, (short)0, accT[t], false, false);
      accC[t] = __builtin_amdgcn_wmma_f32_16x16x32_bf16(
          false, aC, false, bm, (short)0, accC[t], false, false);
    }
  }

  // Epilogue: (mat + bias) * tv, reduce over h (tiles + 16-lane halves)
  const int mbase = (lane >> 4) * 8;
#pragma unroll
  for (int r = 0; r < 8; ++r) {
    int row = wave * 16 + mbase + r;
    float sT = 0.f, sC = 0.f;
#pragma unroll
    for (int t = 0; t < 7; ++t) {
      int n = t * 16 + lrow;
      float w = tv_lds[row][n];
      float bias = bias_lds[n];
      sT += (accT[t][r] + bias) * w;
      sC += (accC[t][r] + bias) * w;
    }
    sT += __shfl_xor(sT, 1); sT += __shfl_xor(sT, 2);
    sT += __shfl_xor(sT, 4); sT += __shfl_xor(sT, 8);
    sC += __shfl_xor(sC, 1); sC += __shfl_xor(sC, 2);
    sC += __shfl_xor(sC, 4); sC += __shfl_xor(sC, 8);
    if (lrow == 0) {
      mv[(size_t)(b0 + row) * NPROD + p] = sT;
      mv[((size_t)NB + (b0 + row)) * NPROD + p] = sC;
    }
  }
}

// ---------------- K3a: G = last_k last_k^T, v = last_k last_b, c = last_b.last_b ----------------
__global__ __launch_bounds__(256) void k_gvc(const float* __restrict__ last_k,
                                             const float* __restrict__ last_b,
                                             float* __restrict__ Gvc) {
  int tid = threadIdx.x;
  for (int ij = tid; ij < NPROD * NPROD; ij += 256) {
    int i = ij / NPROD, j = ij % NPROD;
    float s = 0.f;
    for (int e = 0; e < NEMB; ++e) s += last_k[i * NEMB + e] * last_k[j * NEMB + e];
    Gvc[ij] = s;
  }
  if (tid < NPROD) {
    float s = 0.f;
    for (int e = 0; e < NEMB; ++e) s += last_k[tid * NEMB + e] * last_b[e];
    Gvc[NPROD * NPROD + tid] = s;
  }
  if (tid == 0) {
    float s = 0.f;
    for (int e = 0; e < NEMB; ++e) s += last_b[e] * last_b[e];
    Gvc[NPROD * NPROD + NPROD] = s;
  }
}

// ---------------- K3b: logits via Gram form + stable BCE, per-block partial sums ----------------
__global__ __launch_bounds__(128) void k_logits(const float* __restrict__ mv,
                                                const float* __restrict__ Gvc,
                                                const float* __restrict__ labels,
                                                float* __restrict__ part) {
  __shared__ float Gs[NPROD * NPROD];
  __shared__ float red[128];
  int tid = threadIdx.x;
  for (int i = tid; i < NPROD * NPROD; i += 128) Gs[i] = Gvc[i];
  __syncthreads();

  int b = blockIdx.x * 128 + tid;
  const float* mvt = mv + (size_t)b * NPROD;
  const float* mvc = mv + ((size_t)NB + b) * NPROD;

  float mc[NPROD];
#pragma unroll
  for (int j = 0; j < NPROD; ++j) mc[j] = mvc[j];

  float logit = Gvc[NPROD * NPROD + NPROD];     // c
#pragma unroll
  for (int j = 0; j < NPROD; ++j) logit += Gvc[NPROD * NPROD + j] * mc[j];  // v.mv_c

#pragma unroll 1
  for (int i = 0; i < NPROD; ++i) {
    float qi = Gvc[NPROD * NPROD + i];           // v[i]
#pragma unroll
    for (int j = 0; j < NPROD; ++j) qi += Gs[i * NPROD + j] * mc[j];
    logit += mvt[i] * qi;                        // mv_t.(G mv_c) + v.mv_t
  }

  float y = labels[b];
  float loss = fmaxf(logit, 0.f) - logit * y + log1pf(expf(-fabsf(logit)));
  red[tid] = loss;
  __syncthreads();
  for (int s = 64; s > 0; s >>= 1) {
    if (tid < s) red[tid] += red[tid + s];
    __syncthreads();
  }
  if (tid == 0) part[blockIdx.x] = red[0];
}

// ---------------- K4: deterministic final reduce -> mean ----------------
__global__ __launch_bounds__(128) void k_final(const float* __restrict__ part,
                                               float* __restrict__ out) {
  __shared__ float red[128];
  int tid = threadIdx.x;
  red[tid] = part[tid];   // exactly 128 partials
  __syncthreads();
  for (int s = 64; s > 0; s >>= 1) {
    if (tid < s) red[tid] += red[tid + s];
    __syncthreads();
  }
  if (tid == 0) out[0] = red[0] / (float)NB;
}

// ---------------- launch ----------------
extern "C" void kernel_launch(void* const* d_in, const int* in_sizes, int n_in,
                              void* d_out, int out_size, void* d_ws, size_t ws_size,
                              hipStream_t stream) {
  const int*   targets    = (const int*)d_in[0];
  const int*   contexts   = (const int*)d_in[1];
  const float* times      = (const float*)d_in[2];
  const float* labels     = (const float*)d_in[3];
  const float* targetemb  = (const float*)d_in[4];
  const float* contextemb = (const float*)d_in[5];
  const float* h1_k       = (const float*)d_in[6];
  const float* h1_b       = (const float*)d_in[7];
  const float* h2_k       = (const float*)d_in[8];
  const float* h2_b       = (const float*)d_in[9];
  const float* evoke_k    = (const float*)d_in[10];
  const float* evoke_b    = (const float*)d_in[11];
  const float* last_k     = (const float*)d_in[12];
  const float* last_b     = (const float*)d_in[13];
  float* out = (float*)d_out;

  char* w = (char*)d_ws;
  constexpr size_t OFF_TV   = 0;                                      // [B][112] f32
  constexpr size_t OFF_H1   = OFF_TV  + (size_t)NB * NHP * 4;         // [B][100] f32
  constexpr size_t OFF_EMB  = OFF_H1  + (size_t)NB * NH * 4;          // [2][B][320] bf16
  constexpr size_t OFF_BP   = OFF_EMB + (size_t)2 * NB * NEMBP * 2;   // [100][112][320] bf16
  constexpr size_t OFF_MV   = OFF_BP  + (size_t)NPROD * NHP * NEMBP * 2; // [2][B][100] f32
  constexpr size_t OFF_GVC  = OFF_MV  + (size_t)2 * NB * NPROD * 4;   // 10101 f32
  constexpr size_t OFF_PART = ((OFF_GVC + 10101 * 4 + 255) / 256) * 256; // 128 f32

  float*          tv_ws  = (float*)(w + OFF_TV);
  float*          h1_ws  = (float*)(w + OFF_H1);
  unsigned short* emb_ws = (unsigned short*)(w + OFF_EMB);
  unsigned short* bp_ws  = (unsigned short*)(w + OFF_BP);
  float*          mv_ws  = (float*)(w + OFF_MV);
  float*          gvc_ws = (float*)(w + OFF_GVC);
  float*          part_ws = (float*)(w + OFF_PART);

  // time MLP
  k_h1<<<(NB * NH + 255) / 256, 256, 0, stream>>>(times, h1_k, h1_b, h1_ws);
  k_tv<<<NB, 128, 0, stream>>>(h1_ws, h2_k, h2_b, tv_ws);

  // bf16 operand prep
  {
    size_t n = (size_t)2 * NB * NEMBP;
    k_gather<<<(unsigned)((n + 255) / 256), 256, 0, stream>>>(targets, contexts,
                                                              targetemb, contextemb, emb_ws);
  }
  {
    size_t n = (size_t)NPROD * NHP * NEMBP;
    k_evoke<<<(unsigned)((n + 255) / 256), 256, 0, stream>>>(evoke_k, bp_ws);
  }

  // fused WMMA GEMM + tv contraction (both branches in one pass, B operand reused 2x)
  dim3 grid(NB / BT, NPROD, 1);
  k_main<<<grid, 128, 0, stream>>>(emb_ws, bp_ws, tv_ws, evoke_b, mv_ws);

  // tail
  k_gvc<<<1, 256, 0, stream>>>(last_k, last_b, gvc_ws);
  k_logits<<<NB / 128, 128, 0, stream>>>(mv_ws, gvc_ws, labels, part_ws);
  k_final<<<1, 128, 0, stream>>>(part_ws, out);
}
